// BFeatVanillaGAT_56934086476269
// MI455X (gfx1250) — compile-verified, matrix operand order checked by reading, not run on hardware
//
#include <hip/hip_runtime.h>

// ---------------------------------------------------------------------------
// BFeatVanillaGAT for MI455X (gfx1250).  All GEMMs run on v_wmma_f32_16x16x32_bf16
// (bf16 operands, f32 accumulate).  B operands are pre-packed into WMMA fragment
// layout so each B fragment is two coalesced global_load_b128 per lane.
// A-loader modes are compile-time template parameters (no inner-loop branching).
// ---------------------------------------------------------------------------

#define NNODE 1024
#define NEDGE 32768
#define DN    512
#define DA    512
#define NHEAD 8
#define DK    64
#define NEGV  (-1e9f)

typedef __bf16 v16bf __attribute__((ext_vector_type(16)));
typedef float  v8f   __attribute__((ext_vector_type(8)));

// ---- bf16 helpers (manual RNE) --------------------------------------------
__device__ __forceinline__ unsigned short f2bf_bits(float f) {
  unsigned u = __float_as_uint(f);
  u += 0x7FFFu + ((u >> 16) & 1u);
  return (unsigned short)(u >> 16);
}
__device__ __forceinline__ unsigned f2bf_pack(float lo, float hi) {
  return (unsigned)f2bf_bits(lo) | ((unsigned)f2bf_bits(hi) << 16);
}
__device__ __forceinline__ float bits2f(unsigned short h) {
  return __uint_as_float(((unsigned)h) << 16);
}
// monotone encode of f32 into u32 for atomicMax-based segment_max
__device__ __forceinline__ unsigned enc_f32(float f) {
  unsigned u = __float_as_uint(f);
  return (u & 0x80000000u) ? ~u : (u | 0x80000000u);
}
__device__ __forceinline__ float dec_f32(unsigned e) {
  unsigned u = (e & 0x80000000u) ? (e & 0x7FFFFFFFu) : ~e;
  return __uint_as_float(u);
}
__host__ __device__ __forceinline__ constexpr int kpat(int i) {
  return i < 4 ? 2 * i : 16 + 2 * (i - 4);   // WMMA 16-bit K-pair pattern
}

// ===========================================================================
// B pre-pack: f32 (K x N, optional transpose) -> bf16 fragment-layout tiles.
// Tile (nb, kt) stores 32 lanes x 16 bf16 contiguously at tile = nb*KT + kt.
// ===========================================================================
__global__ void __launch_bounds__(256) pack_b_kernel(
    const float* __restrict__ B, int ldb, int btrans, int KT,
    unsigned short* __restrict__ PB) {
  int t = blockIdx.x * 256 + threadIdx.x;
  int lane = t & 31, tile = t >> 5;
  int kt = tile % KT, nb = tile / KT;
  int half = lane >> 4, l16 = lane & 15;
  int n = nb * 16 + l16;
  union { uint4 q[2]; unsigned u[8]; } out;
#pragma unroll
  for (int i = 0; i < 8; i++) {
    int kb = kt * 32 + kpat(i) + 8 * half;
    float v0 = btrans ? B[(size_t)n * ldb + kb]     : B[(size_t)kb * ldb + n];
    float v1 = btrans ? B[(size_t)n * ldb + kb + 1] : B[(size_t)(kb + 1) * ldb + n];
    out.u[i] = f2bf_pack(v0, v1);
  }
  uint4* dst = (uint4*)(PB + ((size_t)tile * 32 + lane) * 16);
  dst[0] = out.q[0];
  dst[1] = out.q[1];
}

// ===========================================================================
// Generic WMMA GEMM.  Block = 256 threads (8 waves); wave w covers rows
// m0 = blockIdx.x*128 + 16w, 16 rows x 64 cols (4 accumulators, one A frag).
// ===========================================================================
enum { AM_F32 = 0, AM_BF16 = 1, AM_CAT3 = 2, AM_CAT2 = 3, AM_GATHER = 4, AM_HEAD = 5 };

struct GemmP {
  const void* A;   // primary A
  const void* A2;  // secondary (e for CAT3, agg for CAT2, EP for HEAD)
  const int*  gi;  // src indices (CAT3 / GATHER)
  const int*  gj;  // dst indices (CAT3)
  const unsigned short* PB;   // packed bf16 B
  const float* bias;
  const float* addmat;
  const unsigned char* mask;
  const float* resid;
  void* Out;
  int lda, head, ldadd, ldmask, ldres, ldo, K;
  float scale;
  int relu;
};

template <int AMODE, bool OBF>
__global__ void __launch_bounds__(256) gemm_kernel(GemmP p) {
  const int lane = threadIdx.x & 31, wid = threadIdx.x >> 5;
  const int m0 = blockIdx.x * 128 + wid * 16;
  const int n0 = blockIdx.y * 64;
  const int half = lane >> 4, l16 = lane & 15;
  const int mr = m0 + l16;
  const int KT = p.K >> 5;

  // hoisted per-row base pointers (gather indices read once)
  const float *af0 = nullptr, *af1 = nullptr, *af2 = nullptr;
  const unsigned short *ab0 = nullptr, *ab1 = nullptr;
  if constexpr (AMODE == AM_F32)    af0 = (const float*)p.A + (size_t)mr * p.lda;
  if constexpr (AMODE == AM_BF16)   ab0 = (const unsigned short*)p.A + (size_t)mr * p.lda;
  if constexpr (AMODE == AM_GATHER) af0 = (const float*)p.A + (size_t)p.gi[mr] * 512;
  if constexpr (AMODE == AM_CAT3) {
    af0 = (const float*)p.A  + (size_t)p.gi[mr] * 512;
    af1 = (const float*)p.A2 + (size_t)mr * 512;
    af2 = (const float*)p.A  + (size_t)p.gj[mr] * 512;
  }
  if constexpr (AMODE == AM_CAT2) {
    af0 = (const float*)p.A  + (size_t)mr * 512;
    af1 = (const float*)p.A2 + (size_t)mr * 512;
  }
  if constexpr (AMODE == AM_HEAD) {
    ab0 = (const unsigned short*)p.A  + (size_t)mr * 512 + p.head;
    ab1 = (const unsigned short*)p.A2 + (size_t)mr * 512 + p.head;
  }

  v8f acc[4] = {v8f{}, v8f{}, v8f{}, v8f{}};

  for (int kt = 0; kt < KT; kt++) {
    const int k0 = kt * 32;
    union { v16bf v; unsigned u[8]; } A;

    if constexpr (AMODE == AM_BF16) {
#pragma unroll
      for (int i = 0; i < 8; i++)
        A.u[i] = *(const unsigned*)(ab0 + k0 + kpat(i) + 8 * half);
    } else if constexpr (AMODE == AM_HEAD) {
      const unsigned short* b = (k0 < 64) ? ab0 : ab1;
      const int kl = k0 & 63;
#pragma unroll
      for (int i = 0; i < 8; i++) {
        int kb = kl + kpat(i) + 8 * half;
        unsigned lo = b[(size_t)kb * 8];
        unsigned hi = b[(size_t)(kb + 1) * 8];
        A.u[i] = lo | (hi << 16);
      }
    } else {
      const float* b;
      int kl;
      if constexpr (AMODE == AM_CAT3) {
        int seg = k0 >> 9;
        b = seg == 0 ? af0 : (seg == 1 ? af1 : af2);
        kl = k0 & 511;
      } else if constexpr (AMODE == AM_CAT2) {
        b = (k0 < 512) ? af0 : af1;
        kl = k0 & 511;
      } else {  // AM_F32 / AM_GATHER
        b = af0;
        kl = k0;
      }
#pragma unroll
      for (int i = 0; i < 8; i++) {
        float2 f = *(const float2*)(b + kl + kpat(i) + 8 * half);
        A.u[i] = f2bf_pack(f.x, f.y);
      }
    }

#pragma unroll
    for (int f = 0; f < 4; f++) {
      const unsigned short* pb =
          p.PB + (((size_t)(blockIdx.y * 4 + f) * KT + kt) * 32 + lane) * 16;
      union { v16bf v; uint4 q[2]; } B;
      B.q[0] = ((const uint4*)pb)[0];
      B.q[1] = ((const uint4*)pb)[1];
      acc[f] = __builtin_amdgcn_wmma_f32_16x16x32_bf16(
          false, A.v, false, B.v, (short)0, acc[f], false, false);
    }
  }

#pragma unroll
  for (int f = 0; f < 4; f++) {
    int n = n0 + 16 * f + l16;
#pragma unroll
    for (int r = 0; r < 8; r++) {
      int m = m0 + r + 8 * half;
      float v = acc[f][r] * p.scale;
      if (p.bias)   v += p.bias[n];
      if (p.addmat) v += p.addmat[(size_t)m * p.ldadd + n];
      if (p.mask && !p.mask[(size_t)m * p.ldmask + n]) v = NEGV;
      if (p.resid)  v += p.resid[(size_t)m * p.ldres + n];
      if (p.relu && v < 0.f) v = 0.f;
      if constexpr (OBF) ((unsigned short*)p.Out)[(size_t)m * p.ldo + n] = f2bf_bits(v);
      else               ((float*)p.Out)[(size_t)m * p.ldo + n] = v;
    }
  }
}

// ===========================================================================
// Pairwise distance + self_attn_fc MLP:  D[i][j] = d*same, DW[h][i][j]
// ===========================================================================
__global__ void __launch_bounds__(256) pair_mlp_kernel(
    const float* __restrict__ center, const int* __restrict__ bid,
    const float* f1w, const float* f1b, const float* f1g, const float* f1lb,
    const float* f2w, const float* f2b, const float* f2g, const float* f2lb,
    const float* f3w, const float* f3b,
    float* __restrict__ D, float* __restrict__ DW) {
  __shared__ float W[1608];
  const int tid = threadIdx.x;
  for (int c = tid; c < 128;  c += 256) W[c]        = f1w[c];
  for (int c = tid; c < 32;   c += 256) W[128 + c]  = f1b[c];
  for (int c = tid; c < 32;   c += 256) W[160 + c]  = f1g[c];
  for (int c = tid; c < 32;   c += 256) W[192 + c]  = f1lb[c];
  for (int c = tid; c < 1024; c += 256) W[224 + c]  = f2w[c];
  for (int c = tid; c < 32;   c += 256) W[1248 + c] = f2b[c];
  for (int c = tid; c < 32;   c += 256) W[1280 + c] = f2g[c];
  for (int c = tid; c < 32;   c += 256) W[1312 + c] = f2lb[c];
  for (int c = tid; c < 256;  c += 256) W[1344 + c] = f3w[c];
  for (int c = tid; c < 8;    c += 256) W[1600 + c] = f3b[c];
  __syncthreads();

  const int t = blockIdx.x * 256 + tid;
  const int i = t >> 10, j = t & 1023;
  float dx = center[i * 3 + 0] - center[j * 3 + 0];
  float dy = center[i * 3 + 1] - center[j * 3 + 1];
  float dz = center[i * 3 + 2] - center[j * 3 + 2];
  float d  = sqrtf(dx * dx + dy * dy + dz * dz);
  float same = (bid[i] == bid[j]) ? 1.f : 0.f;
  D[t] = d * same;

  float in4[4] = {dx, dy, dz, d};
  float h1[32], h2[32];
#pragma unroll
  for (int o = 0; o < 32; o++) {
    float s = W[128 + o];
    for (int c = 0; c < 4; c++) s += in4[c] * W[c * 32 + o];
    h1[o] = s > 0.f ? s : 0.f;
  }
  float mu = 0.f; for (int o = 0; o < 32; o++) mu += h1[o]; mu *= (1.f / 32.f);
  float var = 0.f; for (int o = 0; o < 32; o++) { float e = h1[o] - mu; var += e * e; }
  var *= (1.f / 32.f);
  float rs = rsqrtf(var + 1e-5f);
  for (int o = 0; o < 32; o++) h1[o] = (h1[o] - mu) * rs * W[160 + o] + W[192 + o];

#pragma unroll
  for (int o = 0; o < 32; o++) {
    float s = W[1248 + o];
    for (int c = 0; c < 32; c++) s += h1[c] * W[224 + c * 32 + o];
    h2[o] = s > 0.f ? s : 0.f;
  }
  mu = 0.f; for (int o = 0; o < 32; o++) mu += h2[o]; mu *= (1.f / 32.f);
  var = 0.f; for (int o = 0; o < 32; o++) { float e = h2[o] - mu; var += e * e; }
  var *= (1.f / 32.f);
  rs = rsqrtf(var + 1e-5f);
  for (int o = 0; o < 32; o++) h2[o] = (h2[o] - mu) * rs * W[1280 + o] + W[1312 + o];

#pragma unroll
  for (int h = 0; h < 8; h++) {
    float s = W[1600 + h];
    for (int c = 0; c < 32; c++) s += h2[c] * W[1344 + c * 8 + h];
    DW[(size_t)h * NNODE * NNODE + t] = s * same;
  }
}

// ===========================================================================
// knn: per-row 16 smallest of (D + inf on diag), jax tie-break (lower index)
// ===========================================================================
__global__ void __launch_bounds__(256) knn_kernel(
    const float* __restrict__ D, const int* __restrict__ bid,
    unsigned char* __restrict__ FINAL) {
  __shared__ unsigned long long keys[1024];
  __shared__ unsigned long long red[256];
  const int i = blockIdx.x, tid = threadIdx.x;
  for (int j = tid; j < 1024; j += 256) {
    float dn = (j == i) ? __uint_as_float(0x7F800000u) : D[(size_t)i * 1024 + j];
    keys[j] = (((unsigned long long)__float_as_uint(dn)) << 32) | (unsigned)j;
    FINAL[(size_t)i * 1024 + j] = 0;
  }
  __syncthreads();
  for (int r = 0; r < 16; r++) {
    unsigned long long mn = ~0ull;
    for (int j = tid; j < 1024; j += 256) { unsigned long long k = keys[j]; if (k < mn) mn = k; }
    red[tid] = mn;
    __syncthreads();
    for (int s = 128; s > 0; s >>= 1) {
      if (tid < s && red[tid + s] < red[tid]) red[tid] = red[tid + s];
      __syncthreads();
    }
    if (tid == 0) {
      int j = (int)(red[0] & 0xFFFFFFFFull);
      keys[j] = ~0ull;
      if (bid[i] == bid[j]) FINAL[(size_t)i * 1024 + j] = 1;
    }
    __syncthreads();
  }
  if (tid == 0) FINAL[(size_t)i * 1024 + i] = 1;
}

// ===========================================================================
// adjacency (bit-packed), hop masks
// ===========================================================================
__global__ void edge_adj_kernel(const int* __restrict__ src, const int* __restrict__ dst,
                                unsigned* __restrict__ ADJ) {
  int e = blockIdx.x * 256 + threadIdx.x;
  int s = src[e], d = dst[e];
  atomicOr(&ADJ[s * 32 + (d >> 5)], 1u << (d & 31));
  atomicOr(&ADJ[d * 32 + (s >> 5)], 1u << (s & 31));
}
__global__ void hop1_kernel(const unsigned* __restrict__ ADJ, unsigned* __restrict__ HOP1) {
  int idx = blockIdx.x * 256 + threadIdx.x;   // (i, w)
  int i = idx >> 5, w = idx & 31;
  unsigned v = ADJ[idx];
  if (w == (i >> 5)) v |= 1u << (i & 31);
  HOP1[idx] = v;
}
__global__ void __launch_bounds__(256) combine_mask_kernel(
    const unsigned* __restrict__ ADJ, const unsigned* __restrict__ HOP1,
    const unsigned char* __restrict__ FINAL,
    unsigned char* __restrict__ M0, unsigned char* __restrict__ M1) {
  int t = blockIdx.x * 256 + threadIdx.x;
  int i = t >> 10, j = t & 1023;
  unsigned char f = FINAL[t];
  unsigned h1 = (HOP1[i * 32 + (j >> 5)] >> (j & 31)) & 1u;
  unsigned inter = 0;
#pragma unroll
  for (int w = 0; w < 32; w++) inter |= HOP1[i * 32 + w] & ADJ[j * 32 + w];
  M0[t] = (unsigned char)(f & (unsigned char)h1);
  M1[t] = (unsigned char)(f & (unsigned char)(inter != 0u));
}

// ===========================================================================
// row softmax (len 1024), row layernorm (len 512)
// ===========================================================================
__global__ void __launch_bounds__(256) softmax_kernel(float* __restrict__ S) {
  __shared__ float red[256];
  float* p = S + (size_t)blockIdx.x * 1024;
  const int tid = threadIdx.x;
  float mx = -3.4e38f;
  for (int j = tid; j < 1024; j += 256) mx = fmaxf(mx, p[j]);
  red[tid] = mx; __syncthreads();
  for (int s = 128; s > 0; s >>= 1) { if (tid < s) red[tid] = fmaxf(red[tid], red[tid + s]); __syncthreads(); }
  mx = red[0]; __syncthreads();
  float sum = 0.f;
  for (int j = tid; j < 1024; j += 256) { float e = __expf(p[j] - mx); p[j] = e; sum += e; }
  red[tid] = sum; __syncthreads();
  for (int s = 128; s > 0; s >>= 1) { if (tid < s) red[tid] += red[tid + s]; __syncthreads(); }
  float inv = 1.f / red[0];
  for (int j = tid; j < 1024; j += 256) p[j] *= inv;
}

__global__ void __launch_bounds__(256) layernorm_kernel(
    const float* __restrict__ X, const float* __restrict__ g,
    const float* __restrict__ b, float* __restrict__ Y) {
  __shared__ float red[256];
  const float* x = X + (size_t)blockIdx.x * 512;
  float* y = Y + (size_t)blockIdx.x * 512;
  const int tid = threadIdx.x;
  float s = x[tid] + x[tid + 256];
  red[tid] = s; __syncthreads();
  for (int st = 128; st > 0; st >>= 1) { if (tid < st) red[tid] += red[tid + st]; __syncthreads(); }
  float mu = red[0] * (1.f / 512.f); __syncthreads();
  float e0 = x[tid] - mu, e1 = x[tid + 256] - mu;
  red[tid] = e0 * e0 + e1 * e1; __syncthreads();
  for (int st = 128; st > 0; st >>= 1) { if (tid < st) red[tid] += red[tid + st]; __syncthreads(); }
  float rs = rsqrtf(red[0] * (1.f / 512.f) + 1e-5f);
  y[tid]       = e0 * rs * g[tid]       + b[tid];
  y[tid + 256] = e1 * rs * g[tid + 256] + b[tid + 256];
}

// ===========================================================================
// feature-wise softmax over d=64 per (edge, head), msg = prob*v, atomic max agg
// ===========================================================================
__global__ void __launch_bounds__(256) prob_msg_kernel(
    const float* __restrict__ PH, const unsigned short* __restrict__ VP,
    const int* __restrict__ src, int h, unsigned* __restrict__ AGGE) {
  int e = blockIdx.x * 256 + threadIdx.x;
  const float* ph = PH + (size_t)e * 64;
  float mx = -3.4e38f;
  for (int d = 0; d < 64; d++) mx = fmaxf(mx, ph[d]);
  float sum = 0.f;
  for (int d = 0; d < 64; d++) sum += __expf(ph[d] - mx);
  float inv = 1.f / sum;
  int s = src[e];
  for (int d = 0; d < 64; d++) {
    float p = __expf(ph[d] - mx) * inv;
    float v = bits2f(VP[(size_t)e * 512 + d * 8 + h]);
    atomicMax(&AGGE[(size_t)s * 512 + d * 8 + h], enc_f32(p * v));
  }
}
__global__ void agg_init_kernel(unsigned* __restrict__ AGGE) {
  AGGE[blockIdx.x * 256 + threadIdx.x] = 0x007FFFFFu;   // enc(-inf)
}
__global__ void agg_decode_kernel(const unsigned* __restrict__ AGGE, float* __restrict__ AGG) {
  int t = blockIdx.x * 256 + threadIdx.x;
  float f = dec_f32(AGGE[t]);
  AGG[t] = isfinite(f) ? f : 0.f;
}

// ===========================================================================
// host side
// ===========================================================================
static inline GemmP gemm_default() {
  GemmP p;
  p.A = p.A2 = nullptr; p.gi = p.gj = nullptr; p.PB = nullptr;
  p.bias = p.addmat = nullptr; p.mask = nullptr; p.resid = nullptr; p.Out = nullptr;
  p.lda = 0; p.head = 0; p.ldadd = 0; p.ldmask = 0; p.ldres = 0; p.ldo = 0; p.K = 0;
  p.scale = 1.f; p.relu = 0;
  return p;
}
template <int AMODE, bool OBF>
static inline void launch_gemm(hipStream_t st, int M, int Ncols, int K, GemmP p) {
  p.K = K;
  dim3 grid(M / 128, Ncols / 64);
  hipLaunchKernelGGL((gemm_kernel<AMODE, OBF>), grid, dim3(256), 0, st, p);
}
static inline void pack_b(hipStream_t st, const float* B, int ldb, int btrans,
                          int K, int N, unsigned short* PB) {
  int KT = K / 32, NT = N / 16;
  hipLaunchKernelGGL(pack_b_kernel, dim3(KT * NT * 32 / 256), dim3(256), 0, st,
                     B, ldb, btrans, KT, PB);
}

extern "C" void kernel_launch(void* const* d_in, const int* in_sizes, int n_in,
                              void* d_out, int out_size, void* d_ws, size_t ws_size,
                              hipStream_t stream) {
  (void)in_sizes; (void)n_in; (void)out_size; (void)ws_size;
  // ---- inputs ----
  const float* obj    = (const float*)d_in[0];
  const float* efeat  = (const float*)d_in[1];
  const int*   src    = (const int*)d_in[2];
  const int*   dst    = (const int*)d_in[2] + NEDGE;
  const int*   bid    = (const int*)d_in[3];
  const float* center = (const float*)d_in[4];
  const float *wq = (const float*)d_in[5],  *bq = (const float*)d_in[6];
  const float *wk = (const float*)d_in[7],  *bk = (const float*)d_in[8];
  const float *wv = (const float*)d_in[9],  *bv = (const float*)d_in[10];
  const float *wo = (const float*)d_in[11], *bo = (const float*)d_in[12];
  const float *lng = (const float*)d_in[13], *lnb = (const float*)d_in[14];
  const float *f1w = (const float*)d_in[15], *f1b = (const float*)d_in[16];
  const float *f1g = (const float*)d_in[17], *f1lb = (const float*)d_in[18];
  const float *f2w = (const float*)d_in[19], *f2b = (const float*)d_in[20];
  const float *f2g = (const float*)d_in[21], *f2lb = (const float*)d_in[22];
  const float *f3w = (const float*)d_in[23], *f3b = (const float*)d_in[24];
  const float *e1w = (const float*)d_in[25], *e1b = (const float*)d_in[26];
  const float *e2w = (const float*)d_in[27], *e2b = (const float*)d_in[28];
  const float *pqw = (const float*)d_in[29], *pqb = (const float*)d_in[30];
  const float *pew = (const float*)d_in[31], *peb = (const float*)d_in[32];
  const float *pvw = (const float*)d_in[33], *pvb = (const float*)d_in[34];
  const float *a1w = (const float*)d_in[35], *a1b = (const float*)d_in[36];
  const float *a2w = (const float*)d_in[37], *a2b = (const float*)d_in[38];
  const float *p1w = (const float*)d_in[39], *p1b = (const float*)d_in[40];
  const float *p2w = (const float*)d_in[41], *p2b = (const float*)d_in[42];

  // ---- workspace carve-up ----
  char* base = (char*)d_ws;
  size_t off = 0;
  auto carve = [&](size_t bytes) { char* r = base + off; off += (bytes + 255) & ~(size_t)255; return (void*)r; };
  float* D     = (float*)carve((size_t)NNODE * NNODE * 4);
  float* DW    = (float*)carve((size_t)NHEAD * NNODE * NNODE * 4);
  unsigned char* FINAL = (unsigned char*)carve((size_t)NNODE * NNODE);
  unsigned char* M0    = (unsigned char*)carve((size_t)NNODE * NNODE);
  unsigned char* M1    = (unsigned char*)carve((size_t)NNODE * NNODE);
  unsigned* ADJ  = (unsigned*)carve((size_t)NNODE * 32 * 4);
  unsigned* HOP1 = (unsigned*)carve((size_t)NNODE * 32 * 4);
  float* X   = (float*)carve((size_t)NNODE * DN * 4);
  float* XT  = (float*)carve((size_t)NNODE * DN * 4);
  float* Qb  = (float*)carve((size_t)NNODE * DN * 4);
  float* Kb  = (float*)carve((size_t)NNODE * DN * 4);
  float* Vb  = (float*)carve((size_t)NNODE * DN * 4);
  float* CTX = (float*)carve((size_t)NNODE * DN * 4);
  float* SC  = (float*)carve((size_t)NHEAD * NNODE * NNODE * 4);
  unsigned short* H1  = (unsigned short*)carve((size_t)NEDGE * 1024 * 2);
  unsigned short* QP  = (unsigned short*)carve((size_t)NEDGE * 512 * 2);
  unsigned short* EPP = (unsigned short*)carve((size_t)NEDGE * 512 * 2);
  unsigned short* VP  = (unsigned short*)carve((size_t)NEDGE * 512 * 2);
  unsigned short* AH  = (unsigned short*)carve((size_t)NEDGE * 128 * 2);
  float* PH  = (float*)carve((size_t)NEDGE * 64 * 4);
  unsigned* AGGE = (unsigned*)carve((size_t)NNODE * 512 * 4);
  float* AGG = (float*)carve((size_t)NNODE * 512 * 4);
  unsigned short* X2 = (unsigned short*)carve((size_t)NNODE * 1024 * 2);
  // packed bf16 weights (fragment layout)
  auto carveP = [&](size_t elems) { return (unsigned short*)carve(elems * 2); };
  unsigned short* Pwq  = carveP(512 * 512);
  unsigned short* Pwk  = carveP(512 * 512);
  unsigned short* Pwv  = carveP(512 * 512);
  unsigned short* Pwo  = carveP(512 * 512);
  unsigned short* Pe1  = carveP((size_t)1536 * 1024);
  unsigned short* Pe2  = carveP((size_t)1024 * 512);
  unsigned short* Ppq  = carveP(512 * 512);
  unsigned short* Ppe  = carveP(512 * 512);
  unsigned short* Ppv  = carveP(512 * 512);
  unsigned short* Pa1  = carveP(128 * 128);
  unsigned short* Pa2  = carveP(128 * 64);
  unsigned short* Pp1  = carveP((size_t)1024 * 1024);
  unsigned short* Pp2  = carveP((size_t)1024 * 512);
  unsigned short* PKb  = carveP((size_t)64 * 1024);   // per-head K^T, repacked each head
  unsigned short* PVb  = carveP((size_t)1024 * 64);   // per-head V, repacked each head

  float* Ebuf = (float*)d_out + (size_t)NNODE * DN;   // e lives in d_out

  // ---- one-time weight packing ----
  pack_b(stream, wq, 512, 0, 512, 512, Pwq);
  pack_b(stream, wk, 512, 0, 512, 512, Pwk);
  pack_b(stream, wv, 512, 0, 512, 512, Pwv);
  pack_b(stream, wo, 512, 0, 512, 512, Pwo);
  pack_b(stream, e1w, 1024, 0, 1536, 1024, Pe1);
  pack_b(stream, e2w, 512, 0, 1024, 512, Pe2);
  pack_b(stream, pqw, 512, 0, 512, 512, Ppq);
  pack_b(stream, pew, 512, 0, 512, 512, Ppe);
  pack_b(stream, pvw, 512, 0, 512, 512, Ppv);
  pack_b(stream, a1w, 128, 0, 128, 128, Pa1);
  pack_b(stream, a2w, 64, 0, 128, 64, Pa2);
  pack_b(stream, p1w, 1024, 0, 1024, 1024, Pp1);
  pack_b(stream, p2w, 512, 0, 1024, 512, Pp2);

  // ---- preprocessing ----
  hipLaunchKernelGGL(pair_mlp_kernel, dim3(NNODE * NNODE / 256), dim3(256), 0, stream,
                     center, bid, f1w, f1b, f1g, f1lb, f2w, f2b, f2g, f2lb, f3w, f3b, D, DW);
  hipLaunchKernelGGL(knn_kernel, dim3(NNODE), dim3(256), 0, stream, D, bid, FINAL);
  hipMemsetAsync(ADJ, 0, (size_t)NNODE * 32 * 4, stream);
  hipLaunchKernelGGL(edge_adj_kernel, dim3(NEDGE / 256), dim3(256), 0, stream, src, dst, ADJ);
  hipLaunchKernelGGL(hop1_kernel, dim3(NNODE * 32 / 256), dim3(256), 0, stream, ADJ, HOP1);
  hipLaunchKernelGGL(combine_mask_kernel, dim3(NNODE * NNODE / 256), dim3(256), 0, stream,
                     ADJ, HOP1, FINAL, M0, M1);
  hipMemcpyAsync(X, obj, (size_t)NNODE * DN * 4, hipMemcpyDeviceToDevice, stream);
  hipMemcpyAsync(Ebuf, efeat, (size_t)NEDGE * 512 * 4, hipMemcpyDeviceToDevice, stream);

  for (int it = 0; it < 2; it++) {
    const unsigned char* Mx = (it == 0) ? M0 : M1;
    // ---------- MHA ----------
    {
      GemmP p = gemm_default();
      p.A = X; p.lda = DN; p.ldo = DN;
      p.PB = Pwq; p.bias = bq; p.Out = Qb; launch_gemm<AM_F32, false>(stream, NNODE, DN, DN, p);
      p.PB = Pwk; p.bias = bk; p.Out = Kb; launch_gemm<AM_F32, false>(stream, NNODE, DN, DN, p);
      p.PB = Pwv; p.bias = bv; p.Out = Vb; launch_gemm<AM_F32, false>(stream, NNODE, DN, DN, p);
    }
    for (int h = 0; h < NHEAD; h++) {   // scores = Q_h K_h^T / 8 + dw_h, mask
      pack_b(stream, Kb + h * DK, 512, 1, DK, NNODE, PKb);
      GemmP p = gemm_default();
      p.A = Qb + h * DK; p.lda = DN;
      p.PB = PKb;
      p.scale = 0.125f;
      p.addmat = DW + (size_t)h * NNODE * NNODE; p.ldadd = NNODE;
      p.mask = Mx; p.ldmask = NNODE;
      p.Out = SC + (size_t)h * NNODE * NNODE; p.ldo = NNODE;
      launch_gemm<AM_F32, false>(stream, NNODE, NNODE, DK, p);
    }
    hipLaunchKernelGGL(softmax_kernel, dim3(NHEAD * NNODE), dim3(256), 0, stream, SC);
    for (int h = 0; h < NHEAD; h++) {   // ctx_h = att_h @ V_h
      pack_b(stream, Vb + h * DK, 512, 0, NNODE, DK, PVb);
      GemmP p = gemm_default();
      p.A = SC + (size_t)h * NNODE * NNODE; p.lda = NNODE;
      p.PB = PVb;
      p.Out = CTX + h * DK; p.ldo = DN;
      launch_gemm<AM_F32, false>(stream, NNODE, DK, NNODE, p);
    }
    {
      GemmP p = gemm_default();           // out proj + residual
      p.A = CTX; p.lda = DN;
      p.PB = Pwo; p.bias = bo;
      p.resid = X; p.ldres = DN;
      p.Out = XT; p.ldo = DN;
      launch_gemm<AM_F32, false>(stream, NNODE, DN, DN, p);
    }
    hipLaunchKernelGGL(layernorm_kernel, dim3(NNODE), dim3(256), 0, stream, XT, lng, lnb, X);

    // ---------- GCN ----------
    if (it == 0) {                        // h1 = relu([x_i|e|x_j] @ e1w + b)  (dead at it==1)
      GemmP p = gemm_default();
      p.A = X; p.A2 = Ebuf; p.gi = src; p.gj = dst;
      p.PB = Pe1; p.bias = e1b; p.relu = 1;
      p.Out = H1; p.ldo = 1024;
      launch_gemm<AM_CAT3, true>(stream, NEDGE, 1024, 1536, p);
    }
    {
      GemmP p = gemm_default();           // ep = e @ pew + peb  (reads old e)
      p.A = Ebuf; p.lda = 512;
      p.PB = Ppe; p.bias = peb;
      p.Out = EPP; p.ldo = 512;
      launch_gemm<AM_F32, true>(stream, NEDGE, 512, 512, p);
    }
    {
      GemmP p = gemm_default();           // q = x[src] @ pqw + pqb
      p.A = X; p.gi = src;
      p.PB = Ppq; p.bias = pqb;
      p.Out = QP; p.ldo = 512;
      launch_gemm<AM_GATHER, true>(stream, NEDGE, 512, 512, p);
    }
    {
      GemmP p = gemm_default();           // v = x[dst] @ pvw + pvb
      p.A = X; p.gi = dst;
      p.PB = Ppv; p.bias = pvb;
      p.Out = VP; p.ldo = 512;
      launch_gemm<AM_GATHER, true>(stream, NEDGE, DA, 512, p);
    }
    if (it == 0) {                        // e_new = h1 @ e2w + e2b  -> overwrites e in d_out
      GemmP p = gemm_default();
      p.A = H1; p.lda = 1024;
      p.PB = Pe2; p.bias = e2b;
      p.Out = Ebuf; p.ldo = 512;
      launch_gemm<AM_BF16, false>(stream, NEDGE, 512, 1024, p);
    }
    hipLaunchKernelGGL(agg_init_kernel, dim3(NNODE * 512 / 256), dim3(256), 0, stream, AGGE);
    for (int h = 0; h < NHEAD; h++) {
      {
        GemmP p = gemm_default();         // Ah = relu(Z_h @ a1w + a1b)
        p.A = QP; p.A2 = EPP; p.head = h;
        p.PB = Pa1; p.bias = a1b; p.relu = 1;
        p.Out = AH; p.ldo = 128;
        launch_gemm<AM_HEAD, true>(stream, NEDGE, 128, 128, p);
      }
      {
        GemmP p = gemm_default();         // Ph = Ah @ a2w + a2b
        p.A = AH; p.lda = 128;
        p.PB = Pa2; p.bias = a2b;
        p.Out = PH; p.ldo = 64;
        launch_gemm<AM_BF16, false>(stream, NEDGE, 64, 128, p);
      }
      hipLaunchKernelGGL(prob_msg_kernel, dim3(NEDGE / 256), dim3(256), 0, stream,
                         PH, VP, src, h, AGGE);
    }
    hipLaunchKernelGGL(agg_decode_kernel, dim3(NNODE * 512 / 256), dim3(256), 0, stream, AGGE, AGG);
    {
      GemmP p = gemm_default();           // h2 = relu([x|agg] @ p1w + p1b)
      p.A = X; p.A2 = AGG;
      p.PB = Pp1; p.bias = p1b; p.relu = 1;
      p.Out = X2; p.ldo = 1024;
      launch_gemm<AM_CAT2, true>(stream, NNODE, 1024, 1024, p);
    }
    {
      GemmP p = gemm_default();           // x = relu(h2 @ p2w + p2b)
      p.A = X2; p.lda = 1024;
      p.PB = Pp2; p.bias = p2b; p.relu = 1;
      p.Out = X; p.ldo = 512;
      launch_gemm<AM_BF16, false>(stream, NNODE, 512, 1024, p);
    }
  }
  hipMemcpyAsync(d_out, X, (size_t)NNODE * DN * 4, hipMemcpyDeviceToDevice, stream);
}
